// XdiffTransformer_13829794693596
// MI455X (gfx1250) — compile-verified, matrix operand order checked by reading, not run on hardware
//
#include <hip/hip_runtime.h>
#include <hip/hip_bf16.h>
#include <math.h>
#include <stdint.h>

#define DEVINL __device__ __forceinline__

typedef unsigned short u16t;
typedef __bf16 bf16_t;
typedef bf16_t bf16x16 __attribute__((ext_vector_type(16)));
typedef float  f32x8   __attribute__((ext_vector_type(8)));

constexpr int B_   = 16;
constexpr int N_   = 512;
constexpr int S_   = 513;   // N+1
constexpr int SP_  = 544;   // padded S: 17*32, 34*16
constexpr int CD_  = 3;
constexpr int FD_  = 8;
constexpr int DM_  = 512;
constexpr int H_   = 8;
constexpr int DFF_ = 2048;
constexpr int L_   = 6;
constexpr int NXF_ = 26;
constexpr int DH_  = 64;
constexpr int PREF_= 512;
constexpr int XWN_ = H_ * NXF_;  // 208
constexpr int XWP_ = 256;        // padded 208 -> 256
constexpr int LDR_ = 40;         // LDS row stride (u16) for a 32-K chunk (80B: 16B-aligned, bank-spread)

// ---------------- scalar helpers ----------------
DEVINL u16t f2bf(float f) {
  unsigned u = __builtin_bit_cast(unsigned, f);
  return (u16t)((u + 0x7FFFu + ((u >> 16) & 1u)) >> 16);
}
DEVINL float bf2f(u16t h) {
  unsigned u = ((unsigned)h) << 16;
  return __builtin_bit_cast(float, u);
}
DEVINL float gelu_f(float x) { return 0.5f * x * (1.0f + erff(x * 0.7071067811865475f)); }
DEVINL float softplus_f(float x) { return x > 0.f ? x + log1pf(__expf(-x)) : log1pf(__expf(x)); }

// ---------------- gfx1250 async copy: global -> LDS (ASYNCcnt-tracked) ----------------
DEVINL void async_ld_b128(const u16t* g, const u16t* l) {
  unsigned lds = (unsigned)(uintptr_t)l;              // low 32 bits of generic ptr = LDS offset
  unsigned long long ga = (unsigned long long)(uintptr_t)g;
  asm volatile("global_load_async_to_lds_b128 %0, %1, off" :: "v"(lds), "v"(ga) : "memory");
}
DEVINL void wait_async0() {
#if __has_builtin(__builtin_amdgcn_s_wait_asynccnt)
  __builtin_amdgcn_s_wait_asynccnt(0);
#else
  asm volatile("s_wait_asynccnt 0x0" ::: "memory");
#endif
}
DEVINL void wait_async3() {
#if __has_builtin(__builtin_amdgcn_s_wait_asynccnt)
  __builtin_amdgcn_s_wait_asynccnt(3);
#else
  asm volatile("s_wait_asynccnt 0x3" ::: "memory");
#endif
}

// ---------------- WMMA fragments ----------------
struct Frag { union { bf16x16 v; uint4 u[2]; }; };

// bf16 A/B fragment loader from row-major storage (rows = M for A, rows = N for B^T)
// lanes 0-15 : row = row0+lane, K = {k0..k0+7 , k0+16..k0+23}
// lanes 16-31: row = row0+lane-16, K = {k0+8..k0+15 , k0+24..k0+31}
DEVINL void ldfrag(Frag& f, const u16t* __restrict__ base, long row0, int stride, int k0, int lane) {
  const int half = (lane >> 4) & 1, lr = lane & 15;
  const u16t* p = base + (long)(row0 + lr) * stride + k0 + half * 8;
  f.u[0] = *(const uint4*)p;
  f.u[1] = *(const uint4*)(p + 16);
}

DEVINL f32x8 wmma_bb(f32x8 c, const Frag& a, const Frag& b) {
  return __builtin_amdgcn_wmma_f32_16x16x32_bf16(false, a.v, false, b.v, (short)0, c, false, false);
}

// ---------------- weight convert: fp32 (K x Nreal) -> bf16 transposed (Npad x K) ----------------
__global__ void k_wconv(const float* __restrict__ w, u16t* __restrict__ o, int K, int Nreal, int Npad) {
  long idx = (long)blockIdx.x * 256 + threadIdx.x;
  if (idx >= (long)Npad * K) return;
  int n = (int)(idx / K), kk = (int)(idx % K);
  float v = (n < Nreal) ? w[(long)kk * Nreal + n] : 0.f;
  o[idx] = f2bf(v);
}

// ---------------- generic WMMA GEMM with async-LDS double-buffered staging ----------------
// C = act(A(MxK) * Bt^T + bias). ACT: 0=none 1=gelu
// OUTM: 0=bf16, 1=f32 store, 2=f32 +=, 3=bf16 transposed V store (B,H,64,SP)
template<int ACT, int OUTM>
__global__ __launch_bounds__(256) void k_gemm(const u16t* __restrict__ A, const u16t* __restrict__ Bt,
                                              const float* __restrict__ bias, void* __restrict__ Cp,
                                              int K, int ldc, int Nreal) {
  __shared__ __align__(16) u16t sA[2][64 * LDR_];
  __shared__ __align__(16) u16t sB[2][128 * LDR_];
  const int tid  = threadIdx.x;
  const int lane = tid & 31, wave = tid >> 5;
  const int half = (lane >> 4) & 1, lr = lane & 15;
  const long mrow0 = (long)blockIdx.x * 64;
  const int  ncol0 = blockIdx.y * 128;
  const int  wm = (wave & 3) * 16;   // wave m-offset inside block tile
  const int  wn = (wave >> 2) * 64;  // wave n-offset inside block tile

  // staging assignment: A = 64 rows x 4 16B-chunks (1/thread); B = 128 rows x 4 chunks (2/thread)
  const int ar = tid >> 2, ac = tid & 3;
  const u16t* gA = A + (mrow0 + ar) * (long)K + ac * 8;
  u16t* lA0 = &sA[0][ar * LDR_ + ac * 8];
  u16t* lA1 = &sA[1][ar * LDR_ + ac * 8];
  int br[2], bc[2];
  const u16t* gB[2]; u16t* lB0[2]; u16t* lB1[2];
#pragma unroll
  for (int j = 0; j < 2; ++j) {
    int idx = tid + j * 256;
    br[j] = idx >> 2; bc[j] = idx & 3;
    gB[j] = Bt + (ncol0 + br[j]) * (long)K + bc[j] * 8;
    lB0[j] = &sB[0][br[j] * LDR_ + bc[j] * 8];
    lB1[j] = &sB[1][br[j] * LDR_ + bc[j] * 8];
  }

  f32x8 acc[4];
#pragma unroll
  for (int t = 0; t < 4; ++t)
#pragma unroll
    for (int i = 0; i < 8; ++i) acc[t][i] = 0.f;

  const int nk = K / 32;
  // prologue: stage 0 into buffer 0
  async_ld_b128(gA, lA0);
  async_ld_b128(gB[0], lB0[0]);
  async_ld_b128(gB[1], lB0[1]);

  for (int kc = 0; kc < nk; ++kc) {
    const int buf = kc & 1;
    if (kc + 1 < nk) {
      const int k2 = (kc + 1) * 32;
      if (buf == 0) {
        async_ld_b128(gA + k2, lA1);
        async_ld_b128(gB[0] + k2, lB1[0]);
        async_ld_b128(gB[1] + k2, lB1[1]);
      } else {
        async_ld_b128(gA + k2, lA0);
        async_ld_b128(gB[0] + k2, lB0[0]);
        async_ld_b128(gB[1] + k2, lB0[1]);
      }
      wait_async3();   // current stage complete (done signals are in-order)
    } else {
      wait_async0();
    }
    __syncthreads();

    Frag a;
    {
      const u16t* p = &sA[buf][(wm + lr) * LDR_ + half * 8];
      a.u[0] = *(const uint4*)p;
      a.u[1] = *(const uint4*)(p + 16);
    }
#pragma unroll
    for (int t = 0; t < 4; ++t) {
      Frag b;
      const u16t* p = &sB[buf][(wn + t * 16 + lr) * LDR_ + half * 8];
      b.u[0] = *(const uint4*)p;
      b.u[1] = *(const uint4*)(p + 16);
      acc[t] = wmma_bb(acc[t], a, b);
    }
    __syncthreads();  // all reads of `buf` done before it is refilled next iteration
  }

#pragma unroll
  for (int t = 0; t < 4; ++t) {
#pragma unroll
    for (int i = 0; i < 8; ++i) {
      long row = mrow0 + wm + i + half * 8;
      int  col = ncol0 + wn + t * 16 + lr;
      float v = acc[t][i] + (col < Nreal ? bias[col] : 0.f);
      if (ACT == 1) v = gelu_f(v);
      if (OUTM == 3) {
        // V projection stored directly transposed: (B,H,64,SP) for attention B-fragments
        int bb = (int)(row / SP_), ss = (int)(row - (long)bb * SP_);
        long tidx = (((long)bb * H_ + (col >> 6)) * 64 + (col & 63)) * SP_ + ss;
        ((u16t*)Cp)[tidx] = f2bf(v);
      } else {
        long idx = row * (long)ldc + col;
        if (OUTM == 0)      ((u16t*)Cp)[idx]  = f2bf(v);
        else if (OUTM == 1) ((float*)Cp)[idx] = v;
        else                ((float*)Cp)[idx] += v;
      }
    }
  }
}

// ---------------- build x_all (B,S,3) ----------------
__global__ void k_xall(const float* __restrict__ x, const float* __restrict__ xin, float* __restrict__ xall) {
  long idx = (long)blockIdx.x * 256 + threadIdx.x;
  if (idx >= (long)B_ * S_ * CD_) return;
  int b = (int)(idx / (S_ * CD_));
  int rem = (int)(idx % (S_ * CD_));
  int s = rem / CD_, c = rem % CD_;
  xall[idx] = (s == 0) ? x[b * CD_ + c] : xin[((long)b * N_ + (s - 1)) * CD_ + c];
}

// ---------------- embed layer 0: (B*N,8) @ (8,512) + softplus -> bf16 ----------------
__global__ void k_embed1(const float* __restrict__ inp, const float* __restrict__ w,
                         const float* __restrict__ bias, u16t* __restrict__ o) {
  long idx = (long)blockIdx.x * 256 + threadIdx.x;
  if (idx >= (long)B_ * N_ * DM_) return;
  long row = idx >> 9; int j = (int)(idx & 511);
  const float* ir = inp + row * FD_;
  float s = bias[j];
#pragma unroll
  for (int f = 0; f < FD_; ++f) s += ir[f] * w[f * DM_ + j];
  o[idx] = f2bf(softplus_f(s));
}

// ---------------- assemble lat (B,SP,512) fp32: token row + gelu(embed2) + zero pad ----------------
__global__ void k_asm_lat(const float* __restrict__ tmp, const float* __restrict__ xtok,
                          float* __restrict__ lat) {
  long idx = (long)blockIdx.x * 256 + threadIdx.x;
  if (idx >= (long)B_ * SP_ * DM_) return;
  int col = (int)(idx & 511);
  long row = idx >> 9;
  int b = (int)(row / SP_), s = (int)(row % SP_);
  float v;
  if (s == 0)       v = xtok[col];
  else if (s <= N_) v = tmp[((long)b * N_ + (s - 1)) * DM_ + col];
  else              v = 0.f;
  lat[idx] = v;
}

// ---------------- layernorm row kernel: fp32 lat -> bf16 out (pad rows -> 0) ----------------
__global__ void k_ln(const float* __restrict__ lat, const float* __restrict__ g,
                     const float* __restrict__ bb, u16t* __restrict__ o) {
  int row = blockIdx.x, lane = threadIdx.x;
  int s = row % SP_;
  const float* xr = lat + (long)row * DM_;
  u16t* orow = o + (long)row * DM_;
  if (s >= S_) {
#pragma unroll
    for (int i = 0; i < 16; ++i) orow[lane + i * 32] = 0;
    return;
  }
  float v[16]; float sum = 0.f;
#pragma unroll
  for (int i = 0; i < 16; ++i) { v[i] = xr[lane + i * 32]; sum += v[i]; }
#pragma unroll
  for (int off = 1; off < 32; off <<= 1) sum += __shfl_xor(sum, off, 32);
  float mean = sum * (1.0f / DM_);
  float var = 0.f;
#pragma unroll
  for (int i = 0; i < 16; ++i) { v[i] -= mean; var += v[i] * v[i]; }
#pragma unroll
  for (int off = 1; off < 32; off <<= 1) var += __shfl_xor(var, off, 32);
  float inv = rsqrtf(var * (1.0f / DM_) + 1e-6f);
#pragma unroll
  for (int i = 0; i < 16; ++i) {
    int c = lane + i * 32;
    orow[c] = f2bf(v[i] * inv * g[c] + bb[c]);
  }
}

// ---------------- xlog: per (b,q) block computes bias[b,q,k,h] for all k,h ----------------
// xlog layout: (B, SP, S, 8) -- q padded so attention bias addressing is affine in q
__global__ __launch_bounds__(256) void k_xlog(const float* __restrict__ xall,
                                              const u16t* __restrict__ xw,
                                              u16t* __restrict__ xlog) {
  int bq = blockIdx.x;
  int b = bq / S_, q = bq % S_;
  __shared__ float sxw[XWN_];
  __shared__ float sq[4];
  int tid = threadIdx.x;
  const u16t* xwr = xw + ((long)b * SP_ + q) * XWP_;
  if (tid < XWN_) sxw[tid] = bf2f(xwr[tid]);
  if (tid < 3)    sq[tid] = xall[((long)b * S_ + q) * CD_ + tid];
  __syncthreads();
  const float PI_ = 3.14159265358979f;
  const float inv26 = 0.19611613513818404f; // 1/sqrt(26)
  for (int k = tid; k < S_; k += 256) {
    const float* xk = xall + ((long)b * S_ + k) * CD_;
    float d0 = sq[0] - xk[0], d1 = sq[1] - xk[1], d2 = sq[2] - xk[2];
    float ss = d0 * d0 + d1 * d1 + d2 * d2;
    float r = (ss > 0.f) ? sqrtf(ss) : 0.f;
    float feats[NXF_];
    feats[0] = r; feats[1] = r * r;
    float s1, c1;
    sincosf(PI_ * r, &s1, &c1);
    float sk = s1, ck = c1;
    feats[2] = s1; feats[3] = c1;
#pragma unroll
    for (int kf = 2; kf <= 12; ++kf) {
      float sn = s1 * ck + c1 * sk;
      float cn = c1 * ck - s1 * sk;
      sk = sn; ck = cn;
      feats[2 * kf] = sk; feats[2 * kf + 1] = ck;
    }
    uint4 pack; u16t* ov = (u16t*)&pack;
#pragma unroll
    for (int h = 0; h < H_; ++h) {
      float a = 0.f;
      const float* wv = sxw + h * NXF_;
#pragma unroll
      for (int f = 0; f < NXF_; ++f) a += feats[f] * wv[f];
      ov[h] = f2bf(a * inv26);
    }
    *(uint4*)(xlog + (((long)b * SP_ + q) * S_ + k) * 8) = pack;
  }
}

// ---------------- flash attention: one wave per (qtile,h,b) ----------------
__global__ __launch_bounds__(32) void k_attn(const u16t* __restrict__ q, const u16t* __restrict__ kbuf,
                                             const u16t* __restrict__ vT, const u16t* __restrict__ xlog,
                                             u16t* __restrict__ av) {
  const int lane = threadIdx.x;
  const int half = lane >> 4, lr = lane & 15;
  const int qt = blockIdx.x, h = blockIdx.y, b = blockIdx.z;
  const long qrow0 = (long)b * SP_ + qt * 16;

  Frag aQ0, aQ1;
  ldfrag(aQ0, q, qrow0, DM_, h * 64,      lane);
  ldfrag(aQ1, q, qrow0, DM_, h * 64 + 32, lane);

  f32x8 oacc[4];
  float mi[8], li[8];
#pragma unroll
  for (int t = 0; t < 4; ++t)
#pragma unroll
    for (int i = 0; i < 8; ++i) oacc[t][i] = 0.f;
#pragma unroll
  for (int i = 0; i < 8; ++i) { mi[i] = -1e30f; li[i] = 0.f; }

  __shared__ __align__(16) u16t pl[16 * 40];
  const u16t* vtb = vT + ((long)b * H_ + h) * 64 * SP_;
  const int qloc = half * 8;
  // xlog base for this wave's q rows (affine in i: + i*S_*8 elements)
  const u16t* xbase = xlog + (((long)b * SP_ + qt * 16 + qloc) * S_) * 8 + h;

  for (int kc = 0; kc < SP_ / 32; ++kc) {
    f32x8 lg[2];
#pragma unroll
    for (int t = 0; t < 2; ++t) {
#pragma unroll
      for (int i = 0; i < 8; ++i) lg[t][i] = 0.f;
      long krow = (long)b * SP_ + kc * 32 + t * 16;
      Frag bk;
      ldfrag(bk, kbuf, krow, DM_, h * 64,      lane); lg[t] = wmma_bb(lg[t], aQ0, bk);
      ldfrag(bk, kbuf, krow, DM_, h * 64 + 32, lane); lg[t] = wmma_bb(lg[t], aQ1, bk);
    }
    // branchless bias: one base address per t, constant per-row offsets
#pragma unroll
    for (int t = 0; t < 2; ++t) {
      int kkr = kc * 32 + t * 16 + lr;
      int kk = kkr < S_ ? kkr : S_ - 1;
      bool kvalid = kkr < S_;
      const u16t* bp = xbase + (long)kk * 8;
#pragma unroll
      for (int i = 0; i < 8; ++i) {
        float bias = bf2f(bp[(long)i * S_ * 8]);
        float v = fmaf(lg[t][i], 0.125f, bias);
        lg[t][i] = kvalid ? v : -1e30f;
      }
    }
    // online softmax (reductions stay within 16-lane halves via xor<=8)
    float mn[8], al[8];
#pragma unroll
    for (int i = 0; i < 8; ++i) {
      float tm = fmaxf(lg[0][i], lg[1][i]);
#pragma unroll
      for (int off = 1; off <= 8; off <<= 1) tm = fmaxf(tm, __shfl_xor(tm, off, 32));
      mn[i] = fmaxf(mi[i], tm);
      al[i] = __expf(mi[i] - mn[i]);
    }
#pragma unroll
    for (int i = 0; i < 8; ++i) {
      float p0 = __expf(lg[0][i] - mn[i]);
      float p1 = __expf(lg[1][i] - mn[i]);
      lg[0][i] = p0; lg[1][i] = p1;
      float s = p0 + p1;
#pragma unroll
      for (int off = 1; off <= 8; off <<= 1) s += __shfl_xor(s, off, 32);
      li[i] = li[i] * al[i] + s;
      mi[i] = mn[i];
    }
#pragma unroll
    for (int t = 0; t < 4; ++t)
#pragma unroll
      for (int i = 0; i < 8; ++i) oacc[t][i] *= al[i];
    // P tile (16x32) to LDS (row-major, stride 40 keeps 16B alignment)
    __syncthreads();
#pragma unroll
    for (int t = 0; t < 2; ++t)
#pragma unroll
      for (int i = 0; i < 8; ++i)
        pl[(i + qloc) * 40 + t * 16 + lr] = f2bf(lg[t][i]);
    __syncthreads();
    Frag aP;
    {
      const u16t* pp = pl + lr * 40 + half * 8;
      aP.u[0] = *(const uint4*)pp;
      aP.u[1] = *(const uint4*)(pp + 16);
    }
#pragma unroll
    for (int t2 = 0; t2 < 4; ++t2) {
      Frag bv;
      ldfrag(bv, vtb, t2 * 16, SP_, kc * 32, lane);
      oacc[t2] = wmma_bb(oacc[t2], aP, bv);
    }
  }
  // epilogue
#pragma unroll
  for (int t2 = 0; t2 < 4; ++t2) {
#pragma unroll
    for (int i = 0; i < 8; ++i) {
      int row = qt * 16 + i + qloc;
      float ov = oacc[t2][i] / li[i];
      av[((long)b * SP_ + row) * DM_ + h * 64 + t2 * 16 + lr] = f2bf(ov);
    }
  }
}

// ---------------- head ----------------
__global__ void k_head1(const u16t* __restrict__ hb, const float* __restrict__ w,
                        const float* __restrict__ bias, float* __restrict__ o1) {
  int idx = blockIdx.x * 256 + threadIdx.x;
  if (idx >= B_ * PREF_) return;
  int b = idx >> 9, j = idx & 511;
  const u16t* hr = hb + (long)b * SP_ * DM_;  // row s=0 (x-token slot)
  float s = bias[j];
  for (int kk = 0; kk < DM_; ++kk) s += bf2f(hr[kk]) * w[kk * PREF_ + j];
  o1[idx] = gelu_f(s);
}

__global__ void k_head2(const float* __restrict__ o1, const float* __restrict__ w,
                        const float* __restrict__ bias, float* __restrict__ out) {
  int b = blockIdx.x, lane = threadIdx.x;
  float s = 0.f;
  for (int kk = lane; kk < PREF_; kk += 32) s += o1[b * PREF_ + kk] * w[kk];
#pragma unroll
  for (int off = 1; off < 32; off <<= 1) s += __shfl_xor(s, off, 32);
  if (lane == 0) out[b] = s + bias[0];
}

// ---------------- host orchestration ----------------
extern "C" void kernel_launch(void* const* d_in, const int* in_sizes, int n_in,
                              void* d_out, int out_size, void* d_ws, size_t ws_size,
                              hipStream_t stream) {
  (void)out_size; (void)ws_size;
  // locate array inputs by unique size
  int ix = 0, ixi = 1, iin = 2;
  for (int i = 0; i < n_in; ++i) if (in_sizes[i] == B_ * CD_)        { ix = i; break; }
  for (int i = 0; i < n_in; ++i) if (in_sizes[i] == B_ * N_ * CD_)   { ixi = i; break; }
  for (int i = 0; i < n_in; ++i) if (in_sizes[i] == B_ * N_ * FD_)   { iin = i; break; }
  const float* x    = (const float*)d_in[ix];
  const float* xin  = (const float*)d_in[ixi];
  const float* inp  = (const float*)d_in[iin];

  // params = remaining leaves, in JAX sorted-key pytree order
  int pidx[256]; int pc = 0;
  for (int i = 0; i < n_in && pc < 200; ++i)
    if (i != ix && i != ixi && i != iin) pidx[pc++] = i;
  auto PP = [&](int j) -> const float* { return (const float*)d_in[pidx[j]]; };
  enum { LO_F1B=0, LO_F1W, LO_F2B, LO_F2W, LO_KB, LO_KW, LO_LN1B, LO_LN1G, LO_LN2B, LO_LN2G,
         LO_OB, LO_OW, LO_QB, LO_QW, LO_VB, LO_VW, LO_XB, LO_XW };
  const int PL = L_ * 18;
  enum { PF_FINB = 0, PF_FINW, PF_IL0B, PF_IL0W, PF_IL1B, PF_IL1W, PF_LNB, PF_LNG,
         PF_PREB, PF_PREW, PF_XTOK };
  auto PG = [&](int which) -> const float* { return PP(PL + which); };

  // workspace allocator
  uintptr_t cur = (uintptr_t)d_ws;
  auto alloc = [&](size_t bytes) -> void* {
    cur = (cur + 255) & ~(uintptr_t)255;
    void* p = (void*)cur; cur += bytes; return p;
  };

  u16t *wq[L_], *wk[L_], *wv[L_], *wo[L_], *wx[L_], *wf1[L_], *wf2[L_];
  for (int l = 0; l < L_; ++l) {
    wq[l]  = (u16t*)alloc((size_t)DM_  * DM_  * 2);
    wk[l]  = (u16t*)alloc((size_t)DM_  * DM_  * 2);
    wv[l]  = (u16t*)alloc((size_t)DM_  * DM_  * 2);
    wo[l]  = (u16t*)alloc((size_t)DM_  * DM_  * 2);
    wx[l]  = (u16t*)alloc((size_t)XWP_ * DM_  * 2);
    wf1[l] = (u16t*)alloc((size_t)DFF_ * DM_  * 2);
    wf2[l] = (u16t*)alloc((size_t)DM_  * DFF_ * 2);
  }
  u16t* w1T   = (u16t*)alloc((size_t)DM_ * DM_ * 2);
  float* xall = (float*)alloc((size_t)B_ * S_ * CD_ * 4);
  u16t* lat0  = (u16t*)alloc((size_t)B_ * N_ * DM_ * 2);
  float* tmp  = (float*)alloc((size_t)B_ * N_ * DM_ * 4);
  float* lat  = (float*)alloc((size_t)B_ * SP_ * DM_ * 4);
  u16t* hb    = (u16t*)alloc((size_t)B_ * SP_ * DM_ * 2);
  u16t* qb    = (u16t*)alloc((size_t)B_ * SP_ * DM_ * 2);
  u16t* kb    = (u16t*)alloc((size_t)B_ * SP_ * DM_ * 2);
  u16t* avb   = (u16t*)alloc((size_t)B_ * SP_ * DM_ * 2);
  u16t* vTb   = (u16t*)alloc((size_t)B_ * H_ * 64 * SP_ * 2);
  u16t* xwb   = (u16t*)alloc((size_t)B_ * SP_ * XWP_ * 2);
  u16t* ffnb  = (u16t*)alloc((size_t)B_ * SP_ * DFF_ * 2);
  u16t* xlogb = (u16t*)alloc((size_t)B_ * SP_ * S_ * 8 * 2);  // q padded to SP
  float* o1   = (float*)alloc((size_t)B_ * PREF_ * 4);

  auto wconv = [&](const float* w, u16t* o, int K, int Nreal, int Npad) {
    long tot = (long)Npad * K;
    k_wconv<<<dim3((unsigned)((tot + 255) / 256)), dim3(256), 0, stream>>>(w, o, K, Nreal, Npad);
  };
  for (int l = 0; l < L_; ++l) {
    const int lb = l * 18;
    wconv(PP(lb + LO_QW),  wq[l],  DM_,  DM_,  DM_);
    wconv(PP(lb + LO_KW),  wk[l],  DM_,  DM_,  DM_);
    wconv(PP(lb + LO_VW),  wv[l],  DM_,  DM_,  DM_);
    wconv(PP(lb + LO_OW),  wo[l],  DM_,  DM_,  DM_);
    wconv(PP(lb + LO_XW),  wx[l],  DM_,  XWN_, XWP_);
    wconv(PP(lb + LO_F1W), wf1[l], DM_,  DFF_, DFF_);
    wconv(PP(lb + LO_F2W), wf2[l], DFF_, DM_,  DM_);
  }
  wconv(PG(PF_IL1W), w1T, DM_, DM_, DM_);

  // coordinates + embedding
  {
    long t1 = (long)B_ * S_ * CD_;
    k_xall<<<dim3((unsigned)((t1 + 255) / 256)), dim3(256), 0, stream>>>(x, xin, xall);
    long t2 = (long)B_ * N_ * DM_;
    k_embed1<<<dim3((unsigned)((t2 + 255) / 256)), dim3(256), 0, stream>>>(inp, PG(PF_IL0W), PG(PF_IL0B), lat0);
    k_gemm<1, 1><<<dim3(B_ * N_ / 64, DM_ / 128), dim3(256), 0, stream>>>(lat0, w1T, PG(PF_IL1B), tmp, DM_, DM_, DM_);
    long t3 = (long)B_ * SP_ * DM_;
    k_asm_lat<<<dim3((unsigned)((t3 + 255) / 256)), dim3(256), 0, stream>>>(tmp, PG(PF_XTOK), lat);
  }

  const int MR = B_ * SP_;                 // padded row count (8704, /64 = 136)

  for (int l = 0; l < L_; ++l) {
    const int lb = l * 18;
    k_ln<<<dim3(MR), dim3(32), 0, stream>>>(lat, PP(lb + LO_LN1G), PP(lb + LO_LN1B), hb);
    k_gemm<0, 0><<<dim3(MR / 64, DM_ / 128), dim3(256), 0, stream>>>(hb, wq[l], PP(lb + LO_QB), qb, DM_, DM_, DM_);
    k_gemm<0, 0><<<dim3(MR / 64, DM_ / 128), dim3(256), 0, stream>>>(hb, wk[l], PP(lb + LO_KB), kb, DM_, DM_, DM_);
    k_gemm<0, 3><<<dim3(MR / 64, DM_ / 128), dim3(256), 0, stream>>>(hb, wv[l], PP(lb + LO_VB), vTb, DM_, DM_, DM_);
    k_gemm<0, 0><<<dim3(MR / 64, XWP_ / 128), dim3(256), 0, stream>>>(qb, wx[l], PP(lb + LO_XB), xwb, DM_, XWP_, XWN_);
    k_xlog<<<dim3(B_ * S_), dim3(256), 0, stream>>>(xall, xwb, xlogb);
    k_attn<<<dim3(SP_ / 16, H_, B_), dim3(32), 0, stream>>>(qb, kb, vTb, xlogb, avb);
    k_gemm<0, 2><<<dim3(MR / 64, DM_ / 128), dim3(256), 0, stream>>>(avb, wo[l], PP(lb + LO_OB), lat, DM_, DM_, DM_);
    k_ln<<<dim3(MR), dim3(32), 0, stream>>>(lat, PP(lb + LO_LN2G), PP(lb + LO_LN2B), hb);
    k_gemm<1, 0><<<dim3(MR / 64, DFF_ / 128), dim3(256), 0, stream>>>(hb, wf1[l], PP(lb + LO_F1B), ffnb, DM_, DFF_, DFF_);
    k_gemm<0, 2><<<dim3(MR / 64, DM_ / 128), dim3(256), 0, stream>>>(ffnb, wf2[l], PP(lb + LO_F2B), lat, DFF_, DM_, DM_);
  }

  k_ln<<<dim3(MR), dim3(32), 0, stream>>>(lat, PG(PF_LNG), PG(PF_LNB), hb);
  k_head1<<<dim3((B_ * PREF_ + 255) / 256), dim3(256), 0, stream>>>(hb, PG(PF_PREW), PG(PF_PREB), o1);
  k_head2<<<dim3(B_), dim3(32), 0, stream>>>(o1, PG(PF_FINW), PG(PF_FINB), (float*)d_out);
}
